// LstmVariationalAutoEncoder_41901700940084
// MI455X (gfx1250) — matrix operand order repaired and checked by reading, not compiled, not run on hardware
//
#include <hip/hip_runtime.h>
#include <hip/hip_bf16.h>

// ---------------------------------------------------------------------------
// LSTM-VAE for MI455X (gfx1250): all matmuls via v_wmma_f32_16x16x32_bf16.
// Big GEMMs use double-buffered GLOBAL_LOAD_ASYNC_TO_LDS_B128 B-panel staging.
// Sizes: B=16, T=256, D=512, H=1024, L=512, V=32000, BT=4096.
// ---------------------------------------------------------------------------

typedef __attribute__((ext_vector_type(16))) __bf16 bf16x16;
typedef __attribute__((ext_vector_type(8)))  float  floatx8;

union FragAB {
    bf16x16        v;
    unsigned short u[16];
    uint4          q[2];
};

__device__ __forceinline__ unsigned short f2bf(float f) {
    unsigned int u = __float_as_uint(f);
    u += 0x7FFFu + ((u >> 16) & 1u);        // round-to-nearest-even
    return (unsigned short)(u >> 16);
}

__device__ __forceinline__ float sigf(float x) { return 1.0f / (1.0f + __expf(-x)); }

// LDS byte-offset of a pointer into a __shared__ object (AS3 ptrtoint gives the
// offset within the workgroup's LDS allocation, which is what the async
// instruction's VDST operand wants).
__device__ __forceinline__ unsigned int lds_offset(const void* p) {
    return (unsigned int)(unsigned long long)
        (__attribute__((address_space(3))) const void*)p;
}

// Issue one 16-byte async global->LDS copy (ASYNCcnt-tracked).
__device__ __forceinline__ void async_copy_b128(unsigned int lds_off, const void* gptr) {
    unsigned long long ga = (unsigned long long)gptr;
    asm volatile("global_load_async_to_lds_b128 %0, %1, off"
                 :: "v"(lds_off), "v"(ga) : "memory");
}

// A fragment (16x32 bf16 A tile; this code's k order, same as load_frag_B):
// lane = m + 16*g ; lane holds k in [k0+16g, k0+16g+16) contiguous
__device__ __forceinline__ bf16x16 load_frag_A(const unsigned short* A, int lda,
                                               int m_base, int k_base, int lane) {
    int m = m_base + (lane & 15);
    int g = (lane >> 4) & 1;
    const unsigned short* p = A + (size_t)m * lda + k_base;
    FragAB f;
    f.q[0] = *(const uint4*)(p + 16 * g);
    f.q[1] = *(const uint4*)(p + 16 * g + 8);
    return f.v;
}

// B fragment: weight stored row-major [N, K] (i.e. we compute X @ W^T).
// lane = n + 16*g ; lane holds k in [k0+16g, k0+16g+16) contiguous.
__device__ __forceinline__ bf16x16 load_frag_B(const unsigned short* Bw, int ldb,
                                               int n_base, int k_base, int lane) {
    int n = n_base + (lane & 15);
    int g = (lane >> 4) & 1;
    const unsigned short* p = Bw + (size_t)n * ldb + k_base + 16 * g;
    FragAB f;
    f.q[0] = *(const uint4*)(p);
    f.q[1] = *(const uint4*)(p + 8);
    return f.v;
}

// ---------------------------------------------------------------------------
// fp32 -> bf16 conversion
// ---------------------------------------------------------------------------
__global__ __launch_bounds__(256) void conv_bf16_kernel(const float* __restrict__ in,
                                                        unsigned short* __restrict__ out,
                                                        int n) {
    int i = blockIdx.x * 256 + threadIdx.x;
    if (i < n) out[i] = f2bf(in[i]);
}

// Build decoder input (x shifted right by one timestep, first step = 0), bf16.
__global__ __launch_bounds__(256) void shift_bf16_kernel(const float* __restrict__ x,
                                                         unsigned short* __restrict__ out) {
    const int Dk = 512, Tk = 256;
    int i = blockIdx.x * 256 + threadIdx.x;          // over 4096*512
    if (i >= 4096 * Dk) return;
    int row = i / Dk, col = i - row * Dk;
    int t = row & (Tk - 1);
    float v = (t == 0) ? 0.0f : x[(size_t)(row - 1) * Dk + col];
    out[i] = f2bf(v);
}

// ---------------------------------------------------------------------------
// Simple WMMA GEMM (one 16x16 tile per wave). Used for the tiny M=16 GEMMs.
// C[M,N] (f32) = A[M,K](bf16) @ B[N,K](bf16)^T + bias1 + bias2
// ---------------------------------------------------------------------------
__global__ __launch_bounds__(256) void gemm_bf16_kernel(const unsigned short* __restrict__ A,
                                                        const unsigned short* __restrict__ Bw,
                                                        float* __restrict__ C,
                                                        int M, int N, int K,
                                                        const float* __restrict__ bias1,
                                                        const float* __restrict__ bias2) {
    const int lane = threadIdx.x & 31;
    const int wave = threadIdx.x >> 5;
    const int ntiles = N >> 4;
    const int tile = blockIdx.x * 8 + wave;
    if (tile >= (M >> 4) * ntiles) return;           // wave-uniform: EXEC stays all-1s
    const int tm = tile / ntiles;
    const int tn = tile - tm * ntiles;

    floatx8 acc = {};
    for (int k = 0; k < K; k += 32) {
        bf16x16 a = load_frag_A(A, K, tm * 16, k, lane);
        bf16x16 b = load_frag_B(Bw, K, tn * 16, k, lane);
        acc = __builtin_amdgcn_wmma_f32_16x16x32_bf16(false, a, false, b,
                                                      (short)0, acc, false, false);
    }
    const int n = tn * 16 + (lane & 15);
    float bb = 0.0f;
    if (bias1) bb += bias1[n];
    if (bias2) bb += bias2[n];
    const int g = lane >> 4;
#pragma unroll
    for (int r = 0; r < 8; ++r) {
        int m = tm * 16 + r + 8 * g;
        C[(size_t)m * N + n] = acc[r] + bb;
    }
}

// ---------------------------------------------------------------------------
// Staged WMMA GEMM: block = 128(M) x 64(N); 8 waves, wave w owns M-tile w and
// all four N-subtiles (A fragment reused 4x). The 64x32 B-panel per k-step is
// staged into LDS with double-buffered GLOBAL_LOAD_ASYNC_TO_LDS_B128 (each of
// the 256 threads copies 16B), overlapped with the WMMA burst on the previous
// panel. Requires M % 16 == 0 (idle waves allowed), N % 64 == 0, K % 32 == 0.
// ---------------------------------------------------------------------------
__global__ __launch_bounds__(256) void gemm_bf16_staged_kernel(
        const unsigned short* __restrict__ A,
        const unsigned short* __restrict__ Bw,
        float* __restrict__ C,
        int M, int N, int K,
        const float* __restrict__ bias1,
        const float* __restrict__ bias2) {
    __shared__ __align__(16) unsigned short stage[2][64 * 32];

    const int tid  = threadIdx.x;
    const int lane = tid & 31;
    const int wave = tid >> 5;
    const int ln = lane & 15, g = lane >> 4;

    const int tn_base = blockIdx.x * 64;
    const int tm      = blockIdx.y * 128 + wave * 16;
    const bool active = (tm < M);                    // wave-uniform

    // This thread's share of each B-panel fill: row n_loc, 8 k-elements.
    const int n_loc = tid >> 2;                      // 0..63
    const int kq    = tid & 3;                       // 0..3
    const unsigned short* srcB = Bw + (size_t)(tn_base + n_loc) * K + 8 * kq;
    const unsigned int dst0 = lds_offset(&stage[0][n_loc * 32 + 8 * kq]);
    const unsigned int dst1 = lds_offset(&stage[1][n_loc * 32 + 8 * kq]);

    async_copy_b128(dst0, srcB);                     // prime buffer 0 (k = 0)

    floatx8 zero = {};
    floatx8 acc[4];
#pragma unroll
    for (int j = 0; j < 4; ++j) acc[j] = zero;

    const unsigned short* pA = A + (size_t)(tm + ln) * K;   // lane's A row

    int buf = 0;
    for (int k = 0; k < K; k += 32) {
        const bool more = (k + 32 < K);
        if (more) async_copy_b128(buf ? dst0 : dst1, srcB + k + 32);

        // In-order async-load completion: leaving <=1 outstanding guarantees
        // the current panel has landed.
        if (more) asm volatile("s_wait_asynccnt 1" ::: "memory");
        else      asm volatile("s_wait_asynccnt 0" ::: "memory");
        __syncthreads();                             // everyone's fill visible

        if (active) {
            if (more) __builtin_prefetch(pA + k + 32, 0, 0);   // A stream ahead
            bf16x16 a = load_frag_A(A, K, tm, k, lane);
#pragma unroll
            for (int j = 0; j < 4; ++j) {
                FragAB fb;
                const unsigned short* p = &stage[buf][(j * 16 + ln) * 32 + 16 * g];
                fb.q[0] = *(const uint4*)(p);
                fb.q[1] = *(const uint4*)(p + 8);
                acc[j] = __builtin_amdgcn_wmma_f32_16x16x32_bf16(
                    false, a, false, fb.v, (short)0, acc[j], false, false);
            }
        }
        __syncthreads();                             // done reading buf before refill
        buf ^= 1;
    }

    if (active) {
#pragma unroll
        for (int j = 0; j < 4; ++j) {
            const int n = tn_base + j * 16 + ln;
            float bb = 0.0f;
            if (bias1) bb += bias1[n];
            if (bias2) bb += bias2[n];
#pragma unroll
            for (int r = 0; r < 8; ++r) {
                int m = tm + r + 8 * g;
                C[(size_t)m * N + n] = acc[j][r] + bb;
            }
        }
    }
}

// ---------------------------------------------------------------------------
// Encoder LSTM recurrence: single workgroup, 1024 threads = 32 waves.
// h[16,1024] bf16 in LDS; c[16,1024] in per-wave accumulator registers.
// Gx already holds x@Wih^T + bih + bhh, laid out [(b*256+t), 4096].
// Wave w owns h-columns [32w, 32w+32): 2 tiles x 4 gate quadrants = 8 WMMA accs.
// ---------------------------------------------------------------------------
__global__ __launch_bounds__(1024) void enc_lstm_kernel(const float* __restrict__ Gx,
                                                        const unsigned short* __restrict__ Whh,
                                                        unsigned short* __restrict__ hT_b) {
    __shared__ __align__(16) unsigned short h_lds[16 * 1024];
    const int tid = threadIdx.x;
    const int lane = tid & 31;
    const int wave = tid >> 5;
    const int ln = lane & 15, g = lane >> 4;
    const int n0 = wave * 32;

    for (int i = tid; i < 16 * 1024; i += 1024) h_lds[i] = 0;
    __syncthreads();

    floatx8 zero = {};
    floatx8 cfrag[2];
    cfrag[0] = zero; cfrag[1] = zero;

    for (int t = 0; t < 256; ++t) {
        floatx8 acc[8];
#pragma unroll
        for (int i = 0; i < 8; ++i) acc[i] = zero;

        for (int k = 0; k < 1024; k += 32) {
            bf16x16 a = load_frag_A(h_lds, 1024, 0, k, lane);
#pragma unroll
            for (int q = 0; q < 4; ++q)
#pragma unroll
                for (int hh = 0; hh < 2; ++hh) {
                    bf16x16 b = load_frag_B(Whh, 1024, q * 1024 + n0 + hh * 16, k, lane);
                    acc[q * 2 + hh] = __builtin_amdgcn_wmma_f32_16x16x32_bf16(
                        false, a, false, b, (short)0, acc[q * 2 + hh], false, false);
                }
        }
        __syncthreads();   // all waves finished reading h for this step

#pragma unroll
        for (int hh = 0; hh < 2; ++hh) {
#pragma unroll
            for (int r = 0; r < 8; ++r) {
                int m = r + 8 * g;
                size_t grow = ((size_t)(m * 256 + t)) * 4096 + n0 + hh * 16 + ln;
                float iv = acc[0 + hh][r] + Gx[grow];
                float fv = acc[2 + hh][r] + Gx[grow + 1024];
                float gv = acc[4 + hh][r] + Gx[grow + 2048];
                float ov = acc[6 + hh][r] + Gx[grow + 3072];
                float cn = sigf(fv) * cfrag[hh][r] + sigf(iv) * tanhf(gv);
                cfrag[hh][r] = cn;
                float hn = sigf(ov) * tanhf(cn);
                h_lds[m * 1024 + n0 + hh * 16 + ln] = f2bf(hn);
            }
        }
        __syncthreads();   // new h visible before next step's WMMA reads
    }

    for (int i = tid; i < 16 * 1024; i += 1024) hT_b[i] = h_lds[i];
}

// ---------------------------------------------------------------------------
// z = eps * exp(0.5*relu(lv_pre)) + relu(mu_pre); emit z4/mu4/lv4 + z in bf16.
// ---------------------------------------------------------------------------
__global__ __launch_bounds__(256) void z_kernel(const float* __restrict__ mu_pre,
                                                const float* __restrict__ lv_pre,
                                                const float* __restrict__ eps,
                                                float* __restrict__ out_z4,
                                                float* __restrict__ out_mu4,
                                                float* __restrict__ out_lv4,
                                                unsigned short* __restrict__ z_b) {
    int i = blockIdx.x * 256 + threadIdx.x;
    if (i >= 16 * 512) return;
    float mu = fmaxf(mu_pre[i], 0.0f);
    float lv = fmaxf(lv_pre[i], 0.0f);
    float z  = eps[i] * __expf(0.5f * lv) + mu;
    out_mu4[i] = mu;
    out_lv4[i] = lv;
    out_z4[i]  = z;
    z_b[i]     = f2bf(z);
}

// ---------------------------------------------------------------------------
// Decoder LSTM recurrence: h0 = z, c0 = 0, L=512 (gates 2048).
// Wave w owns h-columns [16w, 16w+16): 1 tile x 4 quadrants. Streams ys (bf16).
// ---------------------------------------------------------------------------
__global__ __launch_bounds__(1024) void dec_lstm_kernel(const float* __restrict__ Gx,
                                                        const unsigned short* __restrict__ Whh,
                                                        const unsigned short* __restrict__ z_b,
                                                        unsigned short* __restrict__ ys_b) {
    __shared__ __align__(16) unsigned short h_lds[16 * 512];
    const int tid = threadIdx.x;
    const int lane = tid & 31;
    const int wave = tid >> 5;
    const int ln = lane & 15, g = lane >> 4;
    const int n0 = wave * 16;

    for (int i = tid; i < 16 * 512; i += 1024) h_lds[i] = z_b[i];
    __syncthreads();

    floatx8 zero = {};
    floatx8 cfrag = zero;

    for (int t = 0; t < 256; ++t) {
        floatx8 acc[4];
#pragma unroll
        for (int i = 0; i < 4; ++i) acc[i] = zero;

        for (int k = 0; k < 512; k += 32) {
            bf16x16 a = load_frag_A(h_lds, 512, 0, k, lane);
#pragma unroll
            for (int q = 0; q < 4; ++q) {
                bf16x16 b = load_frag_B(Whh, 512, q * 512 + n0, k, lane);
                acc[q] = __builtin_amdgcn_wmma_f32_16x16x32_bf16(
                    false, a, false, b, (short)0, acc[q], false, false);
            }
        }
        __syncthreads();

#pragma unroll
        for (int r = 0; r < 8; ++r) {
            int m = r + 8 * g;
            size_t grow = ((size_t)(m * 256 + t)) * 2048 + n0 + ln;
            float iv = acc[0][r] + Gx[grow];
            float fv = acc[1][r] + Gx[grow + 512];
            float gv = acc[2][r] + Gx[grow + 1024];
            float ov = acc[3][r] + Gx[grow + 1536];
            float cn = sigf(fv) * cfrag[r] + sigf(iv) * tanhf(gv);
            cfrag[r] = cn;
            float hn = sigf(ov) * tanhf(cn);
            unsigned short hb = f2bf(hn);
            h_lds[m * 512 + n0 + ln] = hb;
            ys_b[((size_t)(m * 256 + t)) * 512 + n0 + ln] = hb;
        }
        __syncthreads();
    }
}

// ---------------------------------------------------------------------------
// In-place row softmax (rows of 32000 in d_out).
// ---------------------------------------------------------------------------
__global__ __launch_bounds__(256) void softmax_kernel(float* __restrict__ logits, int N) {
    __shared__ float red[256];
    const int tid = threadIdx.x;
    float* p = logits + (size_t)blockIdx.x * N;

    float mx = -3.402823466e38f;
    for (int i = tid; i < N; i += 256) mx = fmaxf(mx, p[i]);
    red[tid] = mx; __syncthreads();
    for (int s = 128; s > 0; s >>= 1) { if (tid < s) red[tid] = fmaxf(red[tid], red[tid + s]); __syncthreads(); }
    mx = red[0]; __syncthreads();

    float sum = 0.0f;
    for (int i = tid; i < N; i += 256) { float e = __expf(p[i] - mx); p[i] = e; sum += e; }
    red[tid] = sum; __syncthreads();
    for (int s = 128; s > 0; s >>= 1) { if (tid < s) red[tid] += red[tid + s]; __syncthreads(); }
    float inv = 1.0f / red[0];
    for (int i = tid; i < N; i += 256) p[i] *= inv;
}

// ---------------------------------------------------------------------------
// Host orchestration
// ---------------------------------------------------------------------------
extern "C" void kernel_launch(void* const* d_in, const int* in_sizes, int n_in,
                              void* d_out, int out_size, void* d_ws, size_t ws_size,
                              hipStream_t stream) {
    const int B = 16, T = 256, D = 512, H = 1024, L = 512, V = 32000, BT = B * T;

    const float* x        = (const float*)d_in[0];
    const float* eps      = (const float*)d_in[1];
    const float* enc_Wih  = (const float*)d_in[2];
    const float* enc_Whh  = (const float*)d_in[3];
    const float* enc_bih  = (const float*)d_in[4];
    const float* enc_bhh  = (const float*)d_in[5];
    const float* Wmu      = (const float*)d_in[6];
    const float* bmu      = (const float*)d_in[7];
    const float* Wlv      = (const float*)d_in[8];
    const float* blv      = (const float*)d_in[9];
    const float* dec_Wih  = (const float*)d_in[10];
    const float* dec_Whh  = (const float*)d_in[11];
    const float* dec_bih  = (const float*)d_in[12];
    const float* dec_bhh  = (const float*)d_in[13];
    const float* Wout     = (const float*)d_in[14];
    const float* bout     = (const float*)d_in[15];

    float* out    = (float*)d_out;
    float* recons = out;                                   // [B,T,V]
    float* out_z4 = out + (size_t)BT * V;                  // [B,1,1,L]
    float* out_mu4 = out_z4 + (size_t)B * L;
    float* out_lv4 = out_mu4 + (size_t)B * L;

    // Workspace carve-out
    char* w = (char*)d_ws;
    auto take = [&](size_t bytes) { char* p = w; w += (bytes + 255) & ~(size_t)255; return p; };
    unsigned short* xb     = (unsigned short*)take((size_t)BT * D * 2);
    unsigned short* WihE   = (unsigned short*)take((size_t)4 * H * D * 2);
    unsigned short* WhhE   = (unsigned short*)take((size_t)4 * H * H * 2);
    unsigned short* dxb    = (unsigned short*)take((size_t)BT * D * 2);
    unsigned short* WihD   = (unsigned short*)take((size_t)4 * L * D * 2);
    unsigned short* WhhD   = (unsigned short*)take((size_t)4 * L * L * 2);
    unsigned short* WmuB   = (unsigned short*)take((size_t)L * H * 2);
    unsigned short* WlvB   = (unsigned short*)take((size_t)L * H * 2);
    unsigned short* WoutB  = (unsigned short*)take((size_t)V * L * 2);
    unsigned short* hT_b   = (unsigned short*)take((size_t)B * H * 2);
    unsigned short* z_b    = (unsigned short*)take((size_t)B * L * 2);
    unsigned short* ys_b   = (unsigned short*)take((size_t)BT * L * 2);
    float* Gx_enc = (float*)take((size_t)BT * 4 * H * 4);
    float* Gx_dec = (float*)take((size_t)BT * 4 * L * 4);
    float* mu_pre = (float*)take((size_t)B * L * 4);
    float* lv_pre = (float*)take((size_t)B * L * 4);

    auto conv = [&](const float* src, unsigned short* dst, int n) {
        conv_bf16_kernel<<<(n + 255) / 256, 256, 0, stream>>>(src, dst, n);
    };
    auto gemm_small = [&](const unsigned short* A, const unsigned short* Bw, float* C,
                          int M, int N, int K, const float* b1, const float* b2) {
        int tiles = (M / 16) * (N / 16);
        gemm_bf16_kernel<<<(tiles + 7) / 8, 256, 0, stream>>>(A, Bw, C, M, N, K, b1, b2);
    };
    auto gemm_big = [&](const unsigned short* A, const unsigned short* Bw, float* C,
                        int M, int N, int K, const float* b1, const float* b2) {
        dim3 grid(N / 64, (M + 127) / 128);
        gemm_bf16_staged_kernel<<<grid, 256, 0, stream>>>(A, Bw, C, M, N, K, b1, b2);
    };

    // 1) bf16 conversions of activations and weights
    conv(x, xb, BT * D);
    conv(enc_Wih, WihE, 4 * H * D);
    conv(enc_Whh, WhhE, 4 * H * H);
    conv(dec_Wih, WihD, 4 * L * D);
    conv(dec_Whh, WhhD, 4 * L * L);
    conv(Wmu, WmuB, L * H);
    conv(Wlv, WlvB, L * H);
    conv(Wout, WoutB, V * L);
    shift_bf16_kernel<<<(BT * D + 255) / 256, 256, 0, stream>>>(x, dxb);

    // 2) time-parallel input-gate GEMMs (biases folded in), async-staged
    gemm_big(xb,  WihE, Gx_enc, BT, 4 * H, D, enc_bih, enc_bhh);
    gemm_big(dxb, WihD, Gx_dec, BT, 4 * L, D, dec_bih, dec_bhh);

    // 3) encoder recurrence (one WGP-resident workgroup, WMMA per step)
    enc_lstm_kernel<<<1, 1024, 0, stream>>>(Gx_enc, WhhE, hT_b);

    // 4) latent head
    gemm_small(hT_b, WmuB, mu_pre, B, L, H, bmu, nullptr);
    gemm_small(hT_b, WlvB, lv_pre, B, L, H, blv, nullptr);
    z_kernel<<<(B * L + 255) / 256, 256, 0, stream>>>(mu_pre, lv_pre, eps,
                                                      out_z4, out_mu4, out_lv4, z_b);

    // 5) decoder recurrence
    dec_lstm_kernel<<<1, 1024, 0, stream>>>(Gx_dec, WhhD, z_b, ys_b);

    // 6) output projection straight into d_out (dominant GEMM), in-place softmax
    gemm_big(ys_b, WoutB, recons, BT, V, L, bout, nullptr);
    softmax_kernel<<<BT, 256, 0, stream>>>(recons, V);
}
